// share_net_87333864997074
// MI455X (gfx1250) — compile-verified, matrix operand order checked by reading, not run on hardware
//
#include <hip/hip_runtime.h>
#include <cstdint>
#include <cstddef>

#define DEV __device__ __forceinline__

typedef __bf16 bf16_t;
typedef bf16_t v16bf __attribute__((ext_vector_type(16)));
typedef float  v8f   __attribute__((ext_vector_type(8)));

constexpr int BATCH  = 36864;
constexpr int XCOLS  = 106;
constexpr int CHUNKS = BATCH * 20;     // 737280
constexpr int B9     = BATCH / 9;      // 4096
constexpr int XD_K   = 2763;
constexpr int XD_KP  = 2784;           // padded to multiple of 32
constexpr int XC_K   = 306;
constexpr int XC_KP  = 320;

DEV float lrelu(float v){ return v > 0.f ? v : 0.01f * v; }

DEV unsigned short f2bf(float x){
  unsigned u = __float_as_uint(x);
  u += 0x7FFFu + ((u >> 16) & 1u);     // round-to-nearest-even
  return (unsigned short)(u >> 16);
}

// ---------------- slice tables (column starts of strided slices) -------------
__device__ const int S_K1[4]  = {5,6,10,11};
__device__ const int S_K3[4]  = {7,8,12,13};
__device__ const int S_K5[2]  = {9,14};
__device__ const int S_K7[10] = {5,6,7,8,9,10,11,12,13,14};
__device__ const int S_K2[6]  = {0,1,5,6,10,11};
__device__ const int S_K4[6]  = {2,3,7,8,12,13};
__device__ const int S_K6[3]  = {4,9,14};
__device__ const int S_K8[15] = {0,1,2,3,4,5,6,7,8,9,10,11,12,13,14};
__device__ const int S_18[18] = {13,14,15,16,17,22,23,24,25,26,27,28,29,30,35,36,37,38};
__device__ const int S_27[27] = {0,1,2,3,4,9,10,11,12,
                                 13,14,15,16,17,22,23,24,25,26,27,28,29,30,35,36,37,38};

// circular pad: xa column -> x column
DEV int xa2x(int c){ return (c < 5) ? (95 + c) : (c < 105 ? c - 5 : c - 105); }

template<int D>
DEV void gather_xa(const float* __restrict__ xrow, const int* __restrict__ S,
                   int j, float* out){
#pragma unroll
  for (int k = 0; k < D; ++k){
    int idx = D * j + k;
    int p = idx / 20, t = idx % 20;
    out[k] = xrow[xa2x(S[p] + 5 * t)];
  }
}

template<int D, int H>
DEV float mlp3(const float* in, const float* W1, const float* B1,
               const float* W2, const float* B2,
               const float* W3, const float* B3){
  float h1[H], h2[H];
#pragma unroll
  for (int o = 0; o < H; ++o){
    float s = B1[o];
#pragma unroll
    for (int i = 0; i < D; ++i) s += W1[o * D + i] * in[i];
    h1[o] = lrelu(s);
  }
#pragma unroll
  for (int o = 0; o < H; ++o){
    float s = B2[o];
#pragma unroll
    for (int i = 0; i < H; ++i) s += W2[o * H + i] * h1[i];
    h2[o] = lrelu(s);
  }
  float s = B3[0];
#pragma unroll
  for (int i = 0; i < H; ++i) s += W3[i] * h2[i];
  return lrelu(s);
}

// ---------------------------------- L1 ---------------------------------------
struct L1Args { const float* W[24]; const float* B[24]; };

__global__ __launch_bounds__(256)
void l1_kernel(const float* __restrict__ x, float* __restrict__ l1out, L1Args a){
  int c = blockIdx.x * blockDim.x + threadIdx.x;
  if (c >= CHUNKS) return;
  int b = c / 20, j = c % 20;
  const float* xrow = x + (size_t)b * XCOLS;
  float in[15], o[8];
  gather_xa<4 >(xrow, S_K1, j, in); o[0] = mlp3<4 ,4 >(in, a.W[0], a.B[0], a.W[1], a.B[1], a.W[2], a.B[2]);
  gather_xa<4 >(xrow, S_K3, j, in); o[1] = mlp3<4 ,4 >(in, a.W[3], a.B[3], a.W[4], a.B[4], a.W[5], a.B[5]);
  gather_xa<2 >(xrow, S_K5, j, in); o[2] = mlp3<2 ,2 >(in, a.W[6], a.B[6], a.W[7], a.B[7], a.W[8], a.B[8]);
  gather_xa<10>(xrow, S_K7, j, in); o[3] = mlp3<10,16>(in, a.W[9], a.B[9], a.W[10],a.B[10],a.W[11],a.B[11]);
  gather_xa<6 >(xrow, S_K2, j, in); o[4] = mlp3<6 ,8 >(in, a.W[12],a.B[12],a.W[13],a.B[13],a.W[14],a.B[14]);
  gather_xa<6 >(xrow, S_K4, j, in); o[5] = mlp3<6 ,8 >(in, a.W[15],a.B[15],a.W[16],a.B[16],a.W[17],a.B[17]);
  gather_xa<3 >(xrow, S_K6, j, in); o[6] = mlp3<3 ,4 >(in, a.W[18],a.B[18],a.W[19],a.B[19],a.W[20],a.B[20]);
  gather_xa<15>(xrow, S_K8, j, in); o[7] = mlp3<15,16>(in, a.W[21],a.B[21],a.W[22],a.B[22],a.W[23],a.B[23]);
#pragma unroll
  for (int m = 0; m < 8; ++m) l1out[(size_t)c * 8 + m] = o[m];
}

// ---------------------------------- L2 ---------------------------------------
// padded xb row value: xb = [5 raw x vals | 8 L1 outputs] per chunk, 20 chunks,
// circular pad 13 cols each side.
DEV float xbp_val(const float* __restrict__ xrow, const float* __restrict__ l1o,
                  int b, int c /*0..285*/){
  int cp = (c < 13) ? (247 + c) : (c < 273 ? c - 13 : c - 273);
  int jc = cp / 13, s = cp % 13;
  if (s < 5) return xrow[5 * jc + s];
  return l1o[(size_t)(b * 20 + jc) * 8 + (s - 5)];
}

template<int D>
DEV void gather_l2(const float* __restrict__ xrow, const float* __restrict__ l1o,
                   const int* __restrict__ S, int b, int j, float* out){
#pragma unroll
  for (int k = 0; k < D; ++k){
    int idx = D * j + k;
    int p = idx / 20, t = idx % 20;
    out[k] = xbp_val(xrow, l1o, b, S[p] + 13 * t);
  }
}

template<int D>
DEV float mlp4_32(const float* in, const float* W1, const float* B1,
                  const float* W2, const float* B2,
                  const float* W3, const float* B3,
                  const float* W4, const float* B4){
  float h1[32], h2[32];
#pragma unroll
  for (int o = 0; o < 32; ++o){
    float s = B1[o];
#pragma unroll
    for (int i = 0; i < D; ++i) s += W1[o * D + i] * in[i];
    h1[o] = lrelu(s);
  }
#pragma unroll
  for (int o = 0; o < 32; ++o){
    float s = B2[o];
#pragma unroll
    for (int i = 0; i < 32; ++i) s += W2[o * 32 + i] * h1[i];
    h2[o] = lrelu(s);
  }
#pragma unroll
  for (int o = 0; o < 32; ++o){
    float s = B3[o];
#pragma unroll
    for (int i = 0; i < 32; ++i) s += W3[o * 32 + i] * h2[i];
    h1[o] = lrelu(s);                       // reuse h1 as h3
  }
  float s = B4[0];
#pragma unroll
  for (int i = 0; i < 32; ++i) s += W4[i] * h1[i];
  return lrelu(s);
}

struct L2Args { const float* W[8]; const float* B[8]; };  // k1 fc1..4, k2 fc1..4

__global__ __launch_bounds__(256)
void l2_kernel(const float* __restrict__ x, const float* __restrict__ l1out,
               float* __restrict__ l2out, L2Args a){
  // stage both MLPs' weights (~23 KB) in LDS
  __shared__ float sw[5794];
  {
    const float* srcs[16] = {a.W[0],a.B[0],a.W[1],a.B[1],a.W[2],a.B[2],a.W[3],a.B[3],
                             a.W[4],a.B[4],a.W[5],a.B[5],a.W[6],a.B[6],a.W[7],a.B[7]};
    const int lens[16] = {576,32,1024,32,1024,32,32,1, 864,32,1024,32,1024,32,32,1};
    int base = 0;
    for (int s = 0; s < 16; ++s){
      for (int i = threadIdx.x; i < lens[s]; i += 256) sw[base + i] = srcs[s][i];
      base += lens[s];
    }
  }
  __syncthreads();
  int c = blockIdx.x * blockDim.x + threadIdx.x;
  if (c >= CHUNKS) return;
  int b = c / 20, j = c % 20;
  const float* xrow = x + (size_t)b * XCOLS;
  float in[27];
  gather_l2<18>(xrow, l1out, S_18, b, j, in);
  float o1 = mlp4_32<18>(in, sw+0,    sw+576,  sw+608,  sw+1632, sw+1664, sw+2688, sw+2720, sw+2752);
  gather_l2<27>(xrow, l1out, S_27, b, j, in);
  float o2 = mlp4_32<27>(in, sw+2753, sw+3617, sw+3649, sw+4673, sw+4705, sw+5729, sw+5761, sw+5793);
  l2out[(size_t)c * 2 + 0] = o1;
  l2out[(size_t)c * 2 + 1] = o2;
}

// ------------------------------- build xc ------------------------------------
__global__ __launch_bounds__(256)
void build_xc_kernel(const float* __restrict__ x, const float* __restrict__ l1out,
                     const float* __restrict__ l2out,
                     float* __restrict__ xc, unsigned short* __restrict__ xcb){
  long idx = (long)blockIdx.x * blockDim.x + threadIdx.x;
  if (idx >= (long)BATCH * XC_KP) return;
  int b = (int)(idx / XC_KP), c = (int)(idx % XC_KP);
  float v = 0.f;
  if (c < XCOLS) v = x[(size_t)b * XCOLS + c];
  else if (c < XC_K){
    int c2 = c - XCOLS, jc = c2 / 10, s = c2 % 10;
    size_t ci = (size_t)(b * 20 + jc);
    v = (s < 8) ? l1out[ci * 8 + s] : l2out[ci * 2 + (s - 8)];
  }
  if (c < XC_K) xc[(size_t)b * XC_K + c] = v;
  xcb[(size_t)b * XC_KP + c] = f2bf(v);
}

// -------------------------- weight fp32 -> bf16 (padded K) -------------------
__global__ __launch_bounds__(256)
void convert_w_kernel(const float* __restrict__ W, unsigned short* __restrict__ Wb,
                      int N, int K, int Kp){
  long idx = (long)blockIdx.x * blockDim.x + threadIdx.x;
  if (idx >= (long)N * Kp) return;
  int n = (int)(idx / Kp), k = (int)(idx % Kp);
  Wb[idx] = (k < K) ? f2bf(W[(size_t)n * K + k]) : (unsigned short)0;
}

// ------------------------------ WMMA GEMM ------------------------------------
// out[M,N] = act(A[M,K]bf16 * W[N,K]bf16^T + bias) (+ res); optional bf16 copy.
// Block tile 128x128, 8 waves in a 4(M) x 2(N) grid; wave tile 32x64 computed
// as 2x4 v_wmma_f32_16x16x32_bf16 per K-step (8 WMMA per 12 b128 frag loads).
union Frag { v16bf v; uint4 q[2]; };

__global__ __launch_bounds__(256)
void gemm_bf16_wmma(const unsigned short* __restrict__ A,
                    const unsigned short* __restrict__ W,
                    const float* __restrict__ bias,
                    const float* __restrict__ res,
                    float* __restrict__ out,
                    unsigned short* __restrict__ obf,
                    int M, int N, int K, int doLrelu){
  const int lane = threadIdx.x & 31;
  const int wave = threadIdx.x >> 5;
  const int wm = wave & 3, wn = wave >> 2;            // 4x2 wave grid
  const int rowW = blockIdx.y * 128 + wm * 32;        // wave tile 32 (M)
  const int colW = blockIdx.x * 128 + wn * 64;        // wave tile 64 (N)
  const int half = lane >> 4;                          // K-half select
  const int lid  = lane & 15;                          // row/col within tile

  const unsigned short* Arow[2] = {
    A + (size_t)(rowW + 0  + lid) * K,
    A + (size_t)(rowW + 16 + lid) * K };
  const unsigned short* Wrow[4] = {
    W + (size_t)(colW + 0  + lid) * K,
    W + (size_t)(colW + 16 + lid) * K,
    W + (size_t)(colW + 32 + lid) * K,
    W + (size_t)(colW + 48 + lid) * K };

  v8f acc[2][4];
#pragma unroll
  for (int i = 0; i < 2; ++i)
#pragma unroll
    for (int j = 0; j < 4; ++j)
#pragma unroll
      for (int r = 0; r < 8; ++r) acc[i][j][r] = 0.f;

  for (int k0 = 0; k0 < K; k0 += 32){
    if (k0 + 32 < K){                 // near-cache prefetch of next K tile
      __builtin_prefetch(Arow[0] + k0 + 32, 0, 3);
      __builtin_prefetch(Wrow[0] + k0 + 32, 0, 3);
      __builtin_prefetch(Wrow[2] + k0 + 32, 0, 3);
    }
    Frag a[2], b[4];
#pragma unroll
    for (int i = 0; i < 2; ++i){
      const uint4* pa = (const uint4*)(Arow[i] + k0 + 8 * half);
      a[i].q[0] = pa[0];                                // K = 8h .. 8h+7
      a[i].q[1] = pa[2];                                // K = 16+8h .. 16+8h+7
    }
#pragma unroll
    for (int j = 0; j < 4; ++j){
      const uint4* pw = (const uint4*)(Wrow[j] + k0 + 8 * half);
      b[j].q[0] = pw[0];
      b[j].q[1] = pw[2];
    }
#pragma unroll
    for (int i = 0; i < 2; ++i)
#pragma unroll
      for (int j = 0; j < 4; ++j)
        acc[i][j] = __builtin_amdgcn_wmma_f32_16x16x32_bf16(
            false, a[i].v, false, b[j].v, (short)0, acc[i][j], false, false);
  }

#pragma unroll
  for (int i = 0; i < 2; ++i){
#pragma unroll
    for (int j = 0; j < 4; ++j){
      int col = colW + 16 * j + lid;
      float bc = bias[col];
#pragma unroll
      for (int r = 0; r < 8; ++r){
        int row = rowW + 16 * i + 8 * half + r;
        float v = acc[i][j][r] + bc;
        if (doLrelu) v = lrelu(v);
        size_t o = (size_t)row * N + col;
        if (res) v += res[o];
        out[o] = v;
        if (obf) obf[o] = f2bf(v);
      }
    }
  }
}

// ------------------------------- L3 fc8 --------------------------------------
__global__ __launch_bounds__(256)
void fc8_kernel(const float* __restrict__ h, const float* __restrict__ W,
                const float* __restrict__ bb, float* __restrict__ hs){
  int row  = blockIdx.x * 8 + (threadIdx.x >> 5);
  int lane = threadIdx.x & 31;
  if (row >= BATCH) return;
  const float* hr = h + (size_t)row * 512;
  float s = 0.f;
#pragma unroll
  for (int t = 0; t < 16; ++t){ int k = lane + 32 * t; s += W[k] * hr[k]; }
#pragma unroll
  for (int off = 16; off > 0; off >>= 1) s += __shfl_xor(s, off, 32);
  if (lane == 0) hs[row] = lrelu(s + bb[0]);
}

// ------------------------------- build xd ------------------------------------
__global__ __launch_bounds__(256)
void build_xd_kernel(const float* __restrict__ xc, const float* __restrict__ hs,
                     float* __restrict__ xd){
  long idx = (long)blockIdx.x * blockDim.x + threadIdx.x;
  if (idx >= (long)B9 * XD_K) return;
  int c = (int)(idx % XD_K);
  int r = (int)(idx / XD_K);
  int j = c / 307, s = c % 307;
  int b = r * 9 + j;
  xd[idx] = (s < XC_K) ? xc[(size_t)b * XC_K + s] : hs[b];
}

// ----------------------------- BatchNorm -------------------------------------
__global__ __launch_bounds__(256)
void bn_stats_kernel(const float* __restrict__ Z, int M, int K,
                     float* __restrict__ mean, float* __restrict__ inv){
  int col = blockIdx.x;
  if (col >= K) return;
  __shared__ float ss[256], s2[256];
  float a = 0.f, b = 0.f;
  for (int r = threadIdx.x; r < M; r += 256){
    float z = Z[(size_t)r * K + col];
    a += z; b += z * z;
  }
  ss[threadIdx.x] = a; s2[threadIdx.x] = b;
  __syncthreads();
  for (int st = 128; st > 0; st >>= 1){
    if ((int)threadIdx.x < st){
      ss[threadIdx.x] += ss[threadIdx.x + st];
      s2[threadIdx.x] += s2[threadIdx.x + st];
    }
    __syncthreads();
  }
  if (threadIdx.x == 0){
    float m = ss[0] / (float)M;
    float v = s2[0] / (float)M - m * m;
    mean[col] = m;
    inv[col]  = rsqrtf(v + 1e-5f);
  }
}

__global__ __launch_bounds__(256)
void bn_apply_kernel(const float* __restrict__ Z, const float* __restrict__ mean,
                     const float* __restrict__ inv, const float* __restrict__ g,
                     const float* __restrict__ beta, unsigned short* __restrict__ Ab,
                     int M, int K, int Kp){
  long idx = (long)blockIdx.x * blockDim.x + threadIdx.x;
  if (idx >= (long)M * Kp) return;
  int r = (int)(idx / Kp), c = (int)(idx % Kp);
  float v = 0.f;
  if (c < K) v = (Z[(size_t)r * K + c] - mean[c]) * inv[c] * g[c] + beta[c];
  Ab[idx] = f2bf(v);
}

// ============================== host launcher ================================
extern "C" void kernel_launch(void* const* d_in, const int* in_sizes, int n_in,
                              void* d_out, int out_size, void* d_ws, size_t ws_size,
                              hipStream_t stream){
  (void)in_sizes; (void)n_in; (void)out_size; (void)ws_size;
  auto F = [&](int i){ return (const float*)d_in[i]; };
  const float* X = F(0);

  // ---- workspace layout (bytes, 256-aligned) ----
  char* wsb = (char*)d_ws;
  size_t off = 0;
  auto alloc = [&](size_t bytes){ size_t o = off; off = (off + bytes + 255) & ~(size_t)255; return o; };
  size_t oXC  = alloc((size_t)BATCH * XC_K * 4);
  size_t oHS  = alloc((size_t)BATCH * 4);
  size_t oXD  = alloc((size_t)B9 * XD_K * 4);
  size_t pool = off;

  // phase 1-2 overlay
  size_t p = pool;
  auto alloc2 = [&](size_t bytes){ size_t o = p; p = (p + bytes + 255) & ~(size_t)255; return o; };
  size_t oL1O = alloc2((size_t)CHUNKS * 8 * 4);
  size_t oL2O = alloc2((size_t)CHUNKS * 2 * 4);
  size_t oXCB = alloc2((size_t)BATCH * XC_KP * 2);
  size_t oH[2]  = { alloc2((size_t)BATCH * 512 * 4), alloc2((size_t)BATCH * 512 * 4) };
  size_t oHB[2] = { alloc2((size_t)BATCH * 512 * 2), alloc2((size_t)BATCH * 512 * 2) };
  size_t oL3W[7];
  oL3W[0] = alloc2((size_t)512 * XC_KP * 2);
  for (int i = 1; i < 7; ++i) oL3W[i] = alloc2((size_t)512 * 512 * 2);

  // phase 4 overlay (reuses pool; L3 buffers dead by then)
  size_t q = pool;
  auto alloc4 = [&](size_t bytes){ size_t o = q; q = (q + bytes + 255) & ~(size_t)255; return o; };
  size_t oMEAN = alloc4((size_t)XD_KP * 4);
  size_t oINV  = alloc4((size_t)XD_KP * 4);
  size_t oABF  = alloc4((size_t)B9 * XD_KP * 2);
  size_t oY[2] = { alloc4((size_t)B9 * 2048 * 4), alloc4((size_t)B9 * 2048 * 4) };
  const int K4[10]  = {XD_K,2048,2048,2048,2048,2048,1024,1024,1024,1024};
  const int Kp4[10] = {XD_KP,2048,2048,2048,2048,2048,1024,1024,1024,1024};
  const int N4[10]  = {2048,2048,2048,2048,2048,1024,1024,1024,1024,1024};
  size_t oL4W[10];
  for (int i = 0; i < 10; ++i) oL4W[i] = alloc4((size_t)N4[i] * Kp4[i] * 2);

  float* xc  = (float*)(wsb + oXC);
  float* hs  = (float*)(wsb + oHS);
  float* xd  = (float*)(wsb + oXD);
  float* l1o = (float*)(wsb + oL1O);
  float* l2o = (float*)(wsb + oL2O);
  unsigned short* xcb = (unsigned short*)(wsb + oXCB);
  float* h[2]  = {(float*)(wsb + oH[0]),  (float*)(wsb + oH[1])};
  unsigned short* hb[2] = {(unsigned short*)(wsb + oHB[0]), (unsigned short*)(wsb + oHB[1])};
  float* bnM = (float*)(wsb + oMEAN);
  float* bnI = (float*)(wsb + oINV);
  unsigned short* Abf = (unsigned short*)(wsb + oABF);
  float* y[2] = {(float*)(wsb + oY[0]), (float*)(wsb + oY[1])};

  auto gemm = [&](const unsigned short* A, const unsigned short* Wb, const float* bias,
                  const float* res, float* out, unsigned short* obf,
                  int M, int N, int K, int doLr){
    dim3 g(N / 128, M / 128);
    gemm_bf16_wmma<<<g, 256, 0, stream>>>(A, Wb, bias, res, out, obf, M, N, K, doLr);
  };
  auto convw = [&](const float* Wsrc, size_t dstOff, int N, int K, int Kp){
    long total = (long)N * Kp;
    convert_w_kernel<<<(unsigned)((total + 255) / 256), 256, 0, stream>>>(
        Wsrc, (unsigned short*)(wsb + dstOff), N, K, Kp);
  };

  // ---- phase 1: L1 + L2 + xc ----
  L1Args a1;
  { const int G[8] = {0,2,4,6,1,3,5,7};                 // k1,k3,k5,k7,k2,k4,k6,k8
    for (int m = 0; m < 8; ++m)
      for (int l = 0; l < 3; ++l){
        a1.W[3*m+l] = F(1 + G[m]*6 + 2*l);
        a1.B[3*m+l] = F(1 + G[m]*6 + 2*l + 1);
      } }
  l1_kernel<<<CHUNKS/256, 256, 0, stream>>>(X, l1o, a1);

  L2Args a2;
  for (int l = 0; l < 4; ++l){
    a2.W[l]   = F(49 + 2*l); a2.B[l]   = F(50 + 2*l);   // l2k1
    a2.W[4+l] = F(57 + 2*l); a2.B[4+l] = F(58 + 2*l);   // l2k2
  }
  l2_kernel<<<CHUNKS/256, 256, 0, stream>>>(X, l1o, l2o, a2);

  build_xc_kernel<<<(unsigned)(((long)BATCH*XC_KP + 255)/256), 256, 0, stream>>>(
      X, l1o, l2o, xc, xcb);

  // ---- phase 2: L3 residual chain (WMMA) ----
  convw(F(65), oL3W[0], 512, XC_K, XC_KP);
  for (int i = 2; i <= 7; ++i) convw(F(65 + 2*(i-1)), oL3W[i-1], 512, 512, 512);

  gemm(xcb, (unsigned short*)(wsb + oL3W[0]), F(66), nullptr, h[0], hb[0],
       BATCH, 512, XC_KP, 1);
  int cur = 0;
  for (int i = 2; i <= 7; ++i){
    gemm(hb[cur], (unsigned short*)(wsb + oL3W[i-1]), F(66 + 2*(i-1)),
         h[cur], h[1-cur], hb[1-cur], BATCH, 512, 512, 1);
    cur = 1 - cur;
  }
  fc8_kernel<<<BATCH/8, 256, 0, stream>>>(h[cur], F(79), F(80), hs);

  // ---- phase 3: xd ----
  build_xd_kernel<<<(unsigned)(((long)B9*XD_K + 255)/256), 256, 0, stream>>>(xc, hs, xd);

  // ---- phase 4: L4 BN + GEMM chain ----
  for (int i = 0; i < 10; ++i) convw(F(81 + 4*i), oL4W[i], N4[i], K4[i], Kp4[i]);

  // fc1
  bn_stats_kernel<<<XD_K, 256, 0, stream>>>(xd, B9, XD_K, bnM, bnI);
  bn_apply_kernel<<<(unsigned)(((long)B9*XD_KP + 255)/256), 256, 0, stream>>>(
      xd, bnM, bnI, F(83), F(84), Abf, B9, XD_K, XD_KP);
  gemm(Abf, (unsigned short*)(wsb + oL4W[0]), F(82), nullptr, y[0], nullptr,
       B9, 2048, XD_KP, 0);

  int yc = 0;
  for (int i = 1; i < 10; ++i){
    const float* src = y[yc];
    float* dst = (i == 9) ? (float*)d_out : y[1 - yc];
    int K = K4[i], Kp = Kp4[i], N = N4[i];
    int hasRes = (i != 5);                               // fc6 (i==5) breaks residual
    bn_stats_kernel<<<K, 256, 0, stream>>>(src, B9, K, bnM, bnI);
    bn_apply_kernel<<<(unsigned)(((long)B9*Kp + 255)/256), 256, 0, stream>>>(
        src, bnM, bnI, F(83 + 4*i), F(84 + 4*i), Abf, B9, K, Kp);
    gemm(Abf, (unsigned short*)(wsb + oL4W[i]), F(82 + 4*i),
         hasRes ? src : nullptr, dst, nullptr, B9, N, Kp, 0);
    yc = 1 - yc;
  }
}
